// HiResPrecipNet_9x_25x_1563368096252
// MI455X (gfx1250) — matrix operand order repaired and checked by reading, not compiled, not run on hardware
//
#include <hip/hip_runtime.h>
#include <hip/hip_bf16.h>
#include <stdint.h>

// ---------------------------------------------------------------------------
// CDNA5 (gfx1250) implementation of HiResPrecipNet forward.
//  - Dense projections: bf16 WMMA GEMM (f32 accumulate), 128x64 block tile,
//    4 v_wmma per wave per K-step. Main K loop has branch-free staging
//    (spatial out-of-range handled by index clamping; only the K tail pads).
//  - K=1 projections (downscale xr) use a dedicated outer-product kernel.
//  - GATv2 edge softmax + mean aggregation fused per (dst,head) wave32,
//    exploiting dst = repeat(arange,deg) (+ appended self loops) structure.
//  - Deterministic two-stage BatchNorm reduction (no float atomics).
// ---------------------------------------------------------------------------

typedef __attribute__((ext_vector_type(16))) __bf16 v16bf;
typedef __attribute__((ext_vector_type(8)))  __bf16 v8bf;
typedef __attribute__((ext_vector_type(8)))  float  v8f;
typedef __attribute__((ext_vector_type(4), aligned(4))) float f4u;  // dword-aligned vec4

#define BM 128
#define BN 64
#define BK 32

// ---------------------------------------------------------------------------
// GEMM: C[M,N] = act( A[M,K] @ W[K,N] + bias[N] )      act: 0=none, 1=relu
// 256 threads = 8 waves. Block tile 128x64; each wave owns a 32x32 slice
// (2 A-fragments x 2 B-fragments -> 4 WMMA per K-step).
// ---------------------------------------------------------------------------
__launch_bounds__(256)
__global__ void gemm_bias_act(const float* __restrict__ A,
                              const float* __restrict__ W,
                              const float* __restrict__ bias,
                              float* __restrict__ C,
                              int M, int N, int K, int act)
{
    __shared__ __bf16 As[BM][BK];   // row-major A tile   (8 KB)
    __shared__ __bf16 Bs[BN][BK];   // W tile transposed  (4 KB)

    const int tid  = threadIdx.x;
    const int lane = tid & 31;
    const int wave = tid >> 5;       // 0..7
    const int wm   = wave & 3;       // M group: rows [wm*32, wm*32+32)
    const int wn   = wave >> 2;      // N group: cols [wn*32, wn*32+32)
    const int m0   = blockIdx.x * BM;
    const int n0   = blockIdx.y * BN;

    v8f acc[2][2] = {};

    // staging coordinates
    const int ar = tid >> 2;          // A rows ar and ar+64
    const int ak = (tid & 3) * 8;     // A k-offset (8 consecutive)
    const int bk = tid >> 3;          // B k row 0..31
    const int bn = (tid & 7) * 8;     // B n-offset (8 consecutive)

    // clamped spatial indices (out-of-range rows/cols only feed lanes whose
    // stores are masked in the epilogue, so garbage there is safe)
    int armA0 = m0 + ar;        if (armA0 >= M) armA0 = M - 1;
    int armA1 = m0 + ar + 64;   if (armA1 >= M) armA1 = M - 1;
    int bcol[8];
#pragma unroll
    for (int t = 0; t < 8; ++t) {
        int gn = n0 + bn + t;
        bcol[t] = (gn < N) ? gn : (N - 1);
    }
    const int prow = (m0 + (tid & 127)) < M ? (m0 + (tid & 127)) : (M - 1);

    const int kmain = K & ~(BK - 1);

    // helper: fragment build + 4 WMMAs (identical for main and tail)
    auto mma_step = [&]() {
        const int mlane = lane & 15;
        const int abase = (lane >> 4) * 8;      // A: K base per half-wave
        const int kb    = (lane >> 4) * 8;      // B: K base per half-wave
        v16bf afr[2], bfr[2];
#pragma unroll
        for (int s = 0; s < 2; ++s) {
            const int row = wm * 32 + s * 16 + mlane;
#pragma unroll
            for (int t = 0; t < 8; ++t) afr[s][t]     = As[row][abase + t];
#pragma unroll
            for (int t = 0; t < 8; ++t) afr[s][8 + t] = As[row][16 + abase + t];
        }
#pragma unroll
        for (int t2 = 0; t2 < 2; ++t2) {
            const int col = wn * 32 + t2 * 16 + mlane;
#pragma unroll
            for (int t = 0; t < 8; ++t) bfr[t2][t] = Bs[col][kb + t];
#pragma unroll
            for (int t = 0; t < 8; ++t) bfr[t2][8 + t] = Bs[col][16 + kb + t];
        }
#pragma unroll
        for (int s = 0; s < 2; ++s)
#pragma unroll
            for (int t2 = 0; t2 < 2; ++t2)
                acc[s][t2] = __builtin_amdgcn_wmma_f32_16x16x32_bf16(
                    false, afr[s], false, bfr[t2], (short)0, acc[s][t2], false, false);
    };

    // ---- main K loop: branch-free staging ----
    for (int k0 = 0; k0 < kmain; k0 += BK) {
        {   // A rows (two per thread), 8 consecutive k's each
            const int rows[2] = { armA0, armA1 };
#pragma unroll
            for (int half = 0; half < 2; ++half) {
                const float* p = A + (size_t)rows[half] * K + k0 + ak;
                f4u v0 = *(const f4u*)p;
                f4u v1 = *(const f4u*)(p + 4);
                v8bf pack;
#pragma unroll
                for (int t = 0; t < 4; ++t) pack[t]     = (__bf16)v0[t];
#pragma unroll
                for (int t = 0; t < 4; ++t) pack[4 + t] = (__bf16)v1[t];
                *reinterpret_cast<v8bf*>(&As[ar + half * 64][ak]) = pack;
            }
        }
        {   // B row bk, 8 clamped columns, stored transposed
            const float* prow_w = W + (size_t)(k0 + bk) * N;
#pragma unroll
            for (int t = 0; t < 8; ++t) Bs[bn + t][bk] = (__bf16)prow_w[bcol[t]];
        }
        // prefetch next K tile (global_prefetch_b8)
        __builtin_prefetch(&A[(size_t)prow * K + k0 + BK], 0, 1);
        __builtin_prefetch(&W[(size_t)(k0 + BK + bk < K ? k0 + BK + bk : K - 1) * N + n0], 0, 1);
        __syncthreads();
        mma_step();
        __syncthreads();
    }

    // ---- K tail (zero-padded along K) ----
    if (kmain < K) {
        const int k0 = kmain;
        {
            const int rows[2] = { armA0, armA1 };
#pragma unroll
            for (int half = 0; half < 2; ++half) {
                v8bf pack;
#pragma unroll
                for (int t = 0; t < 8; ++t) {
                    const int gk = k0 + ak + t;
                    float v = (gk < K) ? A[(size_t)rows[half] * K + gk] : 0.0f;
                    pack[t] = (__bf16)v;
                }
                *reinterpret_cast<v8bf*>(&As[ar + half * 64][ak]) = pack;
            }
        }
        {
            const int gk = k0 + bk;
            const float* prow_w = W + (size_t)(gk < K ? gk : K - 1) * N;
#pragma unroll
            for (int t = 0; t < 8; ++t) {
                float v = (gk < K) ? prow_w[bcol[t]] : 0.0f;
                Bs[bn + t][bk] = (__bf16)v;
            }
        }
        __syncthreads();
        mma_step();
        __syncthreads();
    }

    // ---- epilogue: VGPR r holds M = r + 8*(lane>>4), N = lane&15 ----
#pragma unroll
    for (int s = 0; s < 2; ++s) {
        const int mbase = m0 + wm * 32 + s * 16 + 8 * (lane >> 4);
#pragma unroll
        for (int t2 = 0; t2 < 2; ++t2) {
            const int gn = n0 + wn * 32 + t2 * 16 + (lane & 15);
            if (gn >= N) continue;
            const float bv = bias[gn];
#pragma unroll
            for (int r = 0; r < 8; ++r) {
                const int gm = mbase + r;
                if (gm < M) {
                    float v = acc[s][t2][r] + bv;
                    if (act == 1) v = fmaxf(v, 0.0f);
                    C[(size_t)gm * N + gn] = v;
                }
            }
        }
    }
}

// ---------------------------------------------------------------------------
// Outer product for K=1 projections: out[i][n] = x[i] * w[n] + b[n]
// ---------------------------------------------------------------------------
__global__ void outer_bias(const float* __restrict__ x, const float* __restrict__ w,
                           const float* __restrict__ b, float* __restrict__ out,
                           int M, int N)
{
    const size_t i = (size_t)blockIdx.x * blockDim.x + threadIdx.x;
    if (i >= (size_t)M * N) return;
    const int r = (int)(i / N), c = (int)(i % N);
    out[i] = x[r] * w[c] + b[c];
}

// ---------------------------------------------------------------------------
// Fused GATv2 edge score -> segment softmax -> mean aggregation.
// One wave32 per (dst node, head). H == 2 fixed. Incoming edges of node i are
// esrc[i*deg .. i*deg+deg) plus an optional self loop. No atomics.
// ---------------------------------------------------------------------------
__launch_bounds__(256)
__global__ void gat_attn_mean(const float* __restrict__ xl,
                              const float* __restrict__ xr,
                              const float* __restrict__ att,
                              const float* __restrict__ bout,
                              const int*   __restrict__ esrc,
                              int n_dst, int C, int deg, int self_loop,
                              int relu, float* __restrict__ out)
{
    const int lane = threadIdx.x & 31;
    const int w    = blockIdx.x * 8 + (threadIdx.x >> 5);
    const int dst  = w >> 1;
    const int h    = w & 1;
    if (dst >= n_dst) return;

    const int tot = deg + self_loop;
    int srcs[8];
    for (int j = 0; j < deg; ++j) srcs[j] = esrc[(size_t)dst * deg + j];
    if (self_loop) srcs[deg] = dst;

    const int HC = 2 * C;
    const float* xrrow = xr  + (size_t)dst * HC + h * C;
    const float* attr  = att + h * C;

    float scores[8];
    float smax = -1e30f;
    for (int j = 0; j < tot; ++j) {
        const float* xlrow = xl + (size_t)srcs[j] * HC + h * C;
        float p = 0.0f;
        for (int c = lane; c < C; c += 32) {
            float v = xlrow[c] + xrrow[c];
            v = (v > 0.0f) ? v : 0.2f * v;        // leaky_relu, NEG=0.2
            p += attr[c] * v;
        }
        for (int off = 16; off; off >>= 1) p += __shfl_xor(p, off, 32);
        scores[j] = p;
        smax = fmaxf(smax, p);
    }

    float den = 0.0f;
    for (int j = 0; j < tot; ++j) { scores[j] = __expf(scores[j] - smax); den += scores[j]; }
    const float scale = 1.0f / (fmaxf(den, 1e-16f) * (float)tot);   // softmax + mean

    const float* brow = bout + h * C;
    float* orow = out + (size_t)dst * HC + h * C;
    for (int c = lane; c < C; c += 32) {
        float a = 0.0f;
        for (int j = 0; j < tot; ++j)
            a += scores[j] * xl[(size_t)srcs[j] * HC + h * C + c];
        float v = a * scale + brow[c];
        if (relu) v = fmaxf(v, 0.0f);
        orow[c] = v;
    }
}

// ---------------------------------------------------------------------------
// BatchNorm (training-style batch stats) — deterministic two-stage reduce.
// ---------------------------------------------------------------------------
#define BN_ROWS 512
__launch_bounds__(256)
__global__ void bn_partial(const float* __restrict__ x, int N, int Cf,
                           float* __restrict__ part)
{
    const int t = threadIdx.x;
    const int c = t % Cf;
    const int rstep = 256 / Cf;
    int r    = blockIdx.x * BN_ROWS + t / Cf;
    int rend = min(N, (int)(blockIdx.x + 1) * BN_ROWS);
    float s = 0.0f, s2 = 0.0f;
    for (; r < rend; r += rstep) {
        float v = x[(size_t)r * Cf + c];
        s += v; s2 += v * v;
    }
    part[(size_t)blockIdx.x * 512 + t]       = s;
    part[(size_t)blockIdx.x * 512 + 256 + t] = s2;
}

__global__ void bn_finish(const float* __restrict__ part, int nblk, int Cf,
                          float* __restrict__ sums)
{
    const int c = threadIdx.x;
    if (c >= Cf) return;
    float s = 0.0f, s2 = 0.0f;
    for (int b = 0; b < nblk; ++b) {
        s  += part[(size_t)b * 512 + c]       + part[(size_t)b * 512 + Cf + c];
        s2 += part[(size_t)b * 512 + 256 + c] + part[(size_t)b * 512 + 256 + Cf + c];
    }
    sums[c]      = s;
    sums[Cf + c] = s2;
}

__global__ void bn_apply_relu(const float* __restrict__ x,
                              const float* __restrict__ sums,
                              const float* __restrict__ g,
                              const float* __restrict__ b,
                              int N, int Cf, float* __restrict__ y)
{
    const size_t i = (size_t)blockIdx.x * blockDim.x + threadIdx.x;
    if (i >= (size_t)N * Cf) return;
    const int c = (int)(i % Cf);
    const float invN = 1.0f / (float)N;
    const float mu   = sums[c] * invN;
    const float var  = sums[Cf + c] * invN - mu * mu;
    float v = g[c] * (x[i] - mu) * rsqrtf(var + 1e-5f) + b[c];
    y[i] = fmaxf(v, 0.0f);
}

// ---------------------------------------------------------------------------
// concat([z, h], axis=-1):  h has ch cols, out has ch+1 cols
// ---------------------------------------------------------------------------
__global__ void concat_z(const float* __restrict__ z, const float* __restrict__ h,
                         float* __restrict__ out, int n, int ch)
{
    const int cols = ch + 1;
    const size_t i = (size_t)blockIdx.x * blockDim.x + threadIdx.x;
    if (i >= (size_t)n * cols) return;
    const int r = (int)(i / cols), c = (int)(i % cols);
    out[i] = (c == 0) ? z[r] : h[(size_t)r * ch + (c - 1)];
}

// final linear to 1 output: y[i] = x[i,:] . w + b
__global__ void linear_out(const float* __restrict__ x, const float* __restrict__ w,
                           const float* __restrict__ b, float* __restrict__ y,
                           int n, int k)
{
    const int i = blockIdx.x * blockDim.x + threadIdx.x;
    if (i >= n) return;
    float s = b[0];
    for (int j = 0; j < k; ++j) s += x[(size_t)i * k + j] * w[j];
    y[i] = s;
}

// ---------------------------------------------------------------------------
// Host orchestration
// ---------------------------------------------------------------------------
struct GatP { const float *Wl, *bl, *Wr, *br, *att, *b; };

extern "C" void kernel_launch(void* const* d_in, const int* in_sizes, int n_in,
                              void* d_out, int out_size, void* d_ws, size_t ws_size,
                              hipStream_t stream)
{
    if (n_in < 77) return;   // expect 77 flattened leaves (insertion order)
    (void)in_sizes; (void)out_size; (void)ws_size;

    auto F = [&](int i) { return (const float*)d_in[i]; };

    const float* x_low  = F(0);   // [10000,125]
    const float* x_9x   = F(1);   // [30000,1]
    const float* x_25x  = F(2);   // [60000,1]
    const float* x_high = F(3);   // [80000,1]
    const float* z_std  = F(4);   // [80000,1]
    const int* e1 = (const int*)d_in[5];  // [2,120000] row0 = src
    const int* e2 = (const int*)d_in[6];  // [2,240000]
    const int* e3 = (const int*)d_in[7];  // [2,320000]
    const int* eh = (const int*)d_in[8];  // [2,480000]

    auto gatP = [&](int base) {
        GatP p; p.Wl = F(base); p.bl = F(base + 1); p.Wr = F(base + 2);
        p.br = F(base + 3); p.att = F(base + 4); p.b = F(base + 5); return p;
    };
    const GatP ds1 = gatP(9),  ds2 = gatP(17), ds3 = gatP(25);
    const float *ds1W = F(15), *ds1b = F(16);
    const float *ds2W = F(23), *ds2b = F(24);
    const float *ds3W = F(31), *ds3b = F(32);
    GatP proc[5]; for (int i = 0; i < 5; ++i) proc[i] = gatP(33 + 6 * i);
    const float *bnG[4], *bnB[4];
    for (int i = 0; i < 4; ++i) { bnG[i] = F(63 + 2 * i); bnB[i] = F(64 + 2 * i); }
    const float *l1W = F(71), *l1b = F(72);
    const float *l2W = F(73), *l2b = F(74);
    const float *l3W = F(75), *l3b = F(76);

    // workspace layout (floats)
    float* ws       = (float*)d_ws;
    float* buf_xl   = ws;                      // 15,000,000  (max 60000x250)
    float* buf_xr   = buf_xl  + 15000000;      // 20,000,000  (max 80000x250)
    float* buf_gat  = buf_xr  + 20000000;      // 20,000,000  (max 80000x250)
    float* buf_a0   = buf_gat + 20000000;      // 10,500,000
    float* buf_a1   = buf_a0  + 10500000;      // 10,500,000
    float* buf_bns  = buf_a1  + 10500000;      // 256 (sum/sumsq)
    float* buf_part = buf_bns + 256;           // 157*512

    auto gemm = [&](const float* A, const float* W, const float* bias, float* C,
                    int M, int N, int K, int act) {
        dim3 g((M + BM - 1) / BM, (N + BN - 1) / BN);
        gemm_bias_act<<<g, dim3(256), 0, stream>>>(A, W, bias, C, M, N, K, act);
    };
    auto outer = [&](const float* x, const float* w, const float* b, float* o,
                     int M, int N) {
        size_t tot = (size_t)M * N;
        outer_bias<<<(unsigned)((tot + 255) / 256), 256, 0, stream>>>(x, w, b, o, M, N);
    };
    auto gat = [&](const float* xl, const float* xr, const GatP& p, const int* esrc,
                   int n_dst, int C, int deg, int self, int relu, float* out) {
        int blocks = (n_dst * 2 + 7) / 8;   // H=2 waves per node, 8 waves/block
        gat_attn_mean<<<blocks, 256, 0, stream>>>(xl, xr, p.att, p.b, esrc,
                                                  n_dst, C, deg, self, relu, out);
    };

    // ---- downscale 1: low[10000,125] -> 9x[30000] ----
    gemm(x_low, ds1.Wl, ds1.bl, buf_xl, 10000, 250, 125, 0);
    outer(x_9x, ds1.Wr, ds1.br, buf_xr, 30000, 250);
    gat(buf_xl, buf_xr, ds1, e1, 30000, 125, 4, 0, 1, buf_gat);
    gemm(buf_gat, ds1W, ds1b, buf_a0, 30000, 125, 250, 1);

    // ---- downscale 2: 9x[30000,125] -> 25x[60000] ----
    gemm(buf_a0, ds2.Wl, ds2.bl, buf_xl, 30000, 250, 125, 0);
    outer(x_25x, ds2.Wr, ds2.br, buf_xr, 60000, 250);
    gat(buf_xl, buf_xr, ds2, e2, 60000, 125, 4, 0, 1, buf_gat);
    gemm(buf_gat, ds2W, ds2b, buf_a1, 60000, 125, 250, 1);

    // ---- downscale 3: 25x[60000,125] -> high[80000] ----
    gemm(buf_a1, ds3.Wl, ds3.bl, buf_xl, 60000, 250, 125, 0);
    outer(x_high, ds3.Wr, ds3.br, buf_xr, 80000, 250);
    gat(buf_xl, buf_xr, ds3, e3, 80000, 125, 4, 0, 1, buf_gat);
    gemm(buf_gat, ds3W, ds3b, buf_a0, 80000, 125, 250, 1);

    // ---- concat z_std with h -> [80000,126] ----
    {
        size_t tot = (size_t)80000 * 126;
        concat_z<<<(unsigned)((tot + 255) / 256), 256, 0, stream>>>(
            z_std, buf_a0, buf_a1, 80000, 125);
    }

    // ---- processor: 5 GATv2 layers on high graph (deg 6 + self loop) ----
    const int NH = 80000, CF = 128;
    const int nblk = (NH + BN_ROWS - 1) / BN_ROWS;   // 157
    const float* cur = buf_a1;
    float* nxt = buf_a0;
    for (int i = 0; i < 5; ++i) {
        const int Kin = (i == 0) ? 126 : 128;
        gemm(cur, proc[i].Wl, proc[i].bl, buf_xl, NH, CF, Kin, 0);
        gemm(cur, proc[i].Wr, proc[i].br, buf_xr, NH, CF, Kin, 0);
        gat(buf_xl, buf_xr, proc[i], eh, NH, 64, 6, 1, (i == 4) ? 1 : 0, buf_gat);
        if (i < 4) {
            bn_partial<<<nblk, 256, 0, stream>>>(buf_gat, NH, CF, buf_part);
            bn_finish<<<1, CF, 0, stream>>>(buf_part, nblk, CF, buf_bns);
            size_t tot = (size_t)NH * CF;
            bn_apply_relu<<<(unsigned)((tot + 255) / 256), 256, 0, stream>>>(
                buf_gat, buf_bns, bnG[i], bnB[i], NH, CF, nxt);
            float* t = nxt; nxt = (float*)cur; cur = t;
        } else {
            cur = buf_gat;   // relu fused in gat
        }
    }

    // ---- prediction MLP ----
    gemm(cur, l1W, l1b, buf_a0, NH, 64, 128, 1);
    gemm(buf_a0, l2W, l2b, buf_a1, NH, 32, 64, 1);
    linear_out<<<(NH + 255) / 256, 256, 0, stream>>>(buf_a1, l3W, l3b,
                                                     (float*)d_out, NH, 32);
}
